// customRNN_52767968199025
// MI455X (gfx1250) — matrix operand order repaired and checked by reading, not compiled, not run on hardware
//
#include <hip/hip_runtime.h>
#include <stdint.h>

// ---------------- problem constants ----------------
#define T_STEPS 64
#define BATCH   512
#define ISZ     512
#define HSZ     1024
#define G4H     4096   // 4*H

// ---------------- tiling ----------------
#define MBLK 128     // batch rows per block
#define NBLK 32      // h-columns per block (x4 gates)
#define KCH  32      // WMMA K per iteration
#define NCHUNK ((ISZ + HSZ) / KCH)     // 48 fused K chunks
#define LDS_STRIDE 20  // dwords per 32-bf16 row (16 data dwords + pad); conflict-free, 16B-aligned rows

typedef __attribute__((ext_vector_type(16))) __bf16 v16bf;
typedef __attribute__((ext_vector_type(8)))  float  v8f;
typedef __attribute__((ext_vector_type(4)))  int    v4i_;

// ---------------- fp32 -> bf16 helpers ----------------
__device__ __forceinline__ unsigned short f2bf(float f) {
  union { float f; uint32_t u; } v; v.f = f;
  uint32_t u = v.u;
  uint32_t r = u + 0x7fffu + ((u >> 16) & 1u);   // round-to-nearest-even
  return (unsigned short)(r >> 16);
}

__device__ __forceinline__ uint32_t pack2(float a, float b) {
#if __has_builtin(__builtin_amdgcn_cvt_pk_bf16_f32)
  auto r = __builtin_amdgcn_cvt_pk_bf16_f32(a, b);   // single v_cvt_pk_bf16_f32
  return __builtin_bit_cast(uint32_t, r);
#else
  return (uint32_t)f2bf(a) | ((uint32_t)f2bf(b) << 16);
#endif
}

// ---------------- async global -> LDS copy (16 bytes) ----------------
__device__ __forceinline__ void copy16_g2l(const unsigned short* g, uint32_t* lds_dst) {
#if __has_builtin(__builtin_amdgcn_global_load_async_to_lds_b128)
  // Param 0 is AS(1) int4* ("int __vector(4) __device__ *" per clang), param 1 LDS side.
  // generic->AS(1): global generic address == global address (int round-trip).
  // generic->AS(3): low 32 bits of a generic LDS address are the LDS byte offset.
  typedef __attribute__((address_space(1))) v4i_ gv4i;
  typedef __attribute__((address_space(3))) v4i_ lv4i;
  __builtin_amdgcn_global_load_async_to_lds_b128(
      (gv4i*)(uintptr_t)g,
      (lv4i*)(uint32_t)(uintptr_t)lds_dst,
      0, 0);
#else
  const uint4 u = *(const uint4*)g;
  *(uint4*)lds_dst = u;          // rows are 16B aligned (stride 20 dwords, q*4 dword offsets)
#endif
}

__device__ __forceinline__ void wait_async_all() {
#if __has_builtin(__builtin_amdgcn_s_wait_asynccnt)
  __builtin_amdgcn_s_wait_asynccnt(0);
#else
  asm volatile("s_wait_asynccnt 0" ::: "memory");
#endif
}

// ---------------- prep kernels ----------------
__global__ void lstm_cvt_pk_kernel(const float* __restrict__ src,
                                   uint32_t* __restrict__ dst, int npairs) {
  int i = blockIdx.x * blockDim.x + threadIdx.x;
  if (i < npairs) dst[i] = pack2(src[2 * i], src[2 * i + 1]);
}

__global__ void lstm_bias_kernel(const float* __restrict__ a, const float* __restrict__ b,
                                 float* __restrict__ o, int n) {
  int i = blockIdx.x * blockDim.x + threadIdx.x;
  if (i < n) o[i] = a[i] + b[i];
}

__global__ void lstm_init_state_kernel(unsigned short* __restrict__ h,
                                       float* __restrict__ c, int n) {
  int i = blockIdx.x * blockDim.x + threadIdx.x;
  if (i < n) { h[i] = 0; c[i] = 0.0f; }
}

__global__ void lstm_finalize_kernel(const float* __restrict__ hlast,
                                     const float* __restrict__ c,
                                     float* __restrict__ hn,
                                     float* __restrict__ cn, int n) {
  int i = blockIdx.x * blockDim.x + threadIdx.x;
  if (i < n) { hn[i] = hlast[i]; cn[i] = c[i]; }
}

// ---------------- fused LSTM step: gates GEMM (bf16 WMMA) + cell update ----------------
// grid: (HSZ/NBLK, BATCH/MBLK) = (32,4), block: 256 threads (8 waves of 32)
// Per wave: two 16-row stripes x one 16-col subtile x 4 gates = 8 WMMA accumulators.
__global__ __launch_bounds__(256, 2)
void lstm_step_kernel(const unsigned short* __restrict__ xbf,   // [T,B,I] bf16 (pre-converted)
                      int t,
                      const unsigned short* __restrict__ hprev, // [B,H] bf16
                      unsigned short* __restrict__ hnext,       // [B,H] bf16
                      const unsigned short* __restrict__ wih,   // [4H,I] bf16
                      const unsigned short* __restrict__ whh,   // [4H,H] bf16
                      const float* __restrict__ bias,           // [4H] fp32 (b_ih+b_hh)
                      float* __restrict__ cstate,               // [B,H] fp32
                      float* __restrict__ out)                  // [T*B,H] fp32
{
  __shared__ uint32_t ldsA[2][MBLK * LDS_STRIDE];        // 2 x 128 rows x (16+4) dwords = 20 KB
  __shared__ uint32_t ldsB[2][4 * NBLK * LDS_STRIDE];    // 2 x 128 rows x (16+4) dwords = 20 KB

  const int tid  = threadIdx.x;
  const int lane = tid & 31;
  const int wave = tid >> 5;            // 0..7
  const int mq   = wave & 3;            // which 32-row pair of the 128-row tile
  const int nsub = wave >> 2;           // which 16-col half of the 32-col tile

  const int n0 = blockIdx.x * NBLK;     // h-column base
  const int m0 = blockIdx.y * MBLK;     // batch row base

  const unsigned short* xt = xbf + ((size_t)t * BATCH) * ISZ;

  // stage one K-chunk (32 x bf16 wide) into LDS buffer `buf` via async copies:
  // A: 128 rows x 4 x 16B ; B: (4 gates x 32 cols) rows x 4 x 16B  -> 4 copies/thread
  auto stage = [&](int buf, int kk) {
    const unsigned short* asrc; size_t astr; int ak;
    const unsigned short* bsrc; size_t bstr; int bk;
    if (kk < ISZ) { asrc = xt;    astr = ISZ; ak = kk;        bsrc = wih; bstr = ISZ; bk = kk; }
    else          { asrc = hprev; astr = HSZ; ak = kk - ISZ;  bsrc = whh; bstr = HSZ; bk = kk - ISZ; }
    uint32_t* bufA = &ldsA[buf][0];
    uint32_t* bufB = &ldsB[buf][0];
#pragma unroll
    for (int it = 0; it < 2; ++it) {
      const int task = tid + it * 256;          // 0..511
      const int row  = task >> 2;               // 0..127
      const int q    = task & 3;                // 16B quarter of the 64B row
      copy16_g2l(asrc + (size_t)(m0 + row) * astr + (ak + q * 8),
                 &bufA[row * LDS_STRIDE + q * 4]);
      const int gate = row >> 5, col = row & 31;
      copy16_g2l(bsrc + (size_t)(gate * HSZ + n0 + col) * bstr + (bk + q * 8),
                 &bufB[row * LDS_STRIDE + q * 4]);
    }
  };

  v8f acc[2][4];   // [stripe][gate i,f,g,o]
#pragma unroll
  for (int s = 0; s < 2; ++s)
#pragma unroll
    for (int g = 0; g < 4; ++g) acc[s][g] = (v8f){};

  const int hi   = (lane >> 4) & 1;
  const int coll = nsub * 16 + (lane & 15);

  stage(0, 0);                                   // prologue: chunk 0 in flight
  for (int kc = 0; kc < NCHUNK; ++kc) {
    const int buf = kc & 1;
    wait_async_all();                            // my copies for chunk kc (and earlier) done
    __syncthreads();                             // everyone's copies visible; prev buf fully read
    if (kc + 1 < NCHUNK) stage(buf ^ 1, (kc + 1) * KCH);   // overlap next chunk with compute

    // ---- A fragments (16x32 MxK): lanes 0-15 K=0..15, lanes 16-31 K=16..31
    union { uint32_t u[8]; v16bf v; } afr[2];
#pragma unroll
    for (int s = 0; s < 2; ++s) {
      const int m = mq * 32 + s * 16 + (lane & 15);
      const uint32_t* ap = &ldsA[buf][m * LDS_STRIDE];
#pragma unroll
      for (int j = 0; j < 8; ++j) {
        const int kd = hi * 8 + j;
        afr[s].u[j] = ap[kd];
      }
    }

    // ---- B fragments (32x16 KxN): lanes 0-15 K=0..15, lanes 16-31 K=16..31
#pragma unroll
    for (int g = 0; g < 4; ++g) {
      union { uint32_t u[8]; v16bf v; } bfr;
      const uint32_t* bp = &ldsB[buf][(g * 32 + coll) * LDS_STRIDE + hi * 8];
#pragma unroll
      for (int j = 0; j < 8; ++j) bfr.u[j] = bp[j];
#pragma unroll
      for (int s = 0; s < 2; ++s)
        acc[s][g] = __builtin_amdgcn_wmma_f32_16x16x32_bf16(
            false, afr[s].v, false, bfr.v, (short)0, acc[s][g], false, false);
    }
  }

  // ---- epilogue: bias + LSTM cell, fully in-register
  const int colg = n0 + coll;
  const float bi  = bias[0 * HSZ + colg];
  const float bfo = bias[1 * HSZ + colg];
  const float bg  = bias[2 * HSZ + colg];
  const float bo  = bias[3 * HSZ + colg];
#pragma unroll
  for (int s = 0; s < 2; ++s) {
    const int rbase = m0 + mq * 32 + s * 16 + hi * 8;   // C/D layout: V_r -> M=r (+8 hi lanes)
#pragma unroll
    for (int r = 0; r < 8; ++r) {
      const int row = rbase + r;
      const float gi = acc[s][0][r] + bi;
      const float gf = acc[s][1][r] + bfo;
      const float gg = acc[s][2][r] + bg;
      const float go = acc[s][3][r] + bo;
      const float si = 1.0f / (1.0f + __expf(-gi));
      const float sf = 1.0f / (1.0f + __expf(-gf));
      const float tg = tanhf(gg);
      const float so = 1.0f / (1.0f + __expf(-go));
      const size_t idx = (size_t)row * HSZ + colg;
      const float cn = sf * cstate[idx] + si * tg;
      cstate[idx] = cn;
      const float h = so * tanhf(cn);
      out[(size_t)t * BATCH * HSZ + idx] = h;
      hnext[idx] = f2bf(h);
    }
  }
}

// ---------------- host launcher ----------------
extern "C" void kernel_launch(void* const* d_in, const int* in_sizes, int n_in,
                              void* d_out, int out_size, void* d_ws, size_t ws_size,
                              hipStream_t stream) {
  const float* input = (const float*)d_in[0];
  // d_in[1] = ranges, d_in[2] = lengths : unused (matches reference)
  const float* W_ih = (const float*)d_in[3];
  const float* W_hh = (const float*)d_in[4];
  const float* b_ih = (const float*)d_in[5];
  const float* b_hh = (const float*)d_in[6];
  float* out = (float*)d_out;

  // workspace layout (bytes), total ~48 MB
  char* ws = (char*)d_ws;
  unsigned short* xbf   = (unsigned short*)(ws);                 // 33,554,432  [T,B,I] bf16
  unsigned short* wih_b = (unsigned short*)(ws + 33554432);      //  4,194,304  [4H,I] bf16
  unsigned short* whh_b = (unsigned short*)(ws + 37748736);      //  8,388,608  [4H,H] bf16
  float*          bias  = (float*)(ws + 46137344);               //     16,384  [4H] fp32
  unsigned short* h0    = (unsigned short*)(ws + 46153728);      //  1,048,576  [B,H] bf16
  unsigned short* h1    = (unsigned short*)(ws + 47202304);      //  1,048,576  [B,H] bf16
  float*          cst   = (float*)(ws + 48250880);               //  2,097,152  [B,H] fp32

  const int NX = T_STEPS * BATCH * ISZ / 2;   // pairs
  const int NW1 = G4H * ISZ / 2;
  const int NW2 = G4H * HSZ / 2;
  lstm_cvt_pk_kernel<<<(NX  + 255) / 256, 256, 0, stream>>>(input, (uint32_t*)xbf,   NX);
  lstm_cvt_pk_kernel<<<(NW1 + 255) / 256, 256, 0, stream>>>(W_ih,  (uint32_t*)wih_b, NW1);
  lstm_cvt_pk_kernel<<<(NW2 + 255) / 256, 256, 0, stream>>>(W_hh,  (uint32_t*)whh_b, NW2);
  lstm_bias_kernel<<<(G4H + 255) / 256, 256, 0, stream>>>(b_ih, b_hh, bias, G4H);
  lstm_init_state_kernel<<<(BATCH * HSZ + 255) / 256, 256, 0, stream>>>(h0, cst, BATCH * HSZ);

  dim3 grid(HSZ / NBLK, BATCH / MBLK);   // (32, 4) = 128 blocks per step (~1 per WGP)
  for (int t = 0; t < T_STEPS; ++t) {
    const unsigned short* hp = (t & 1) ? h1 : h0;
    unsigned short*       hn = (t & 1) ? h0 : h1;
    lstm_step_kernel<<<grid, 256, 0, stream>>>(xbf, t, hp, hn, wih_b, whh_b, bias, cst, out);
  }

  // outputs: [T*B,H] hs, then h_n [B,H], then c_n [B,H]
  lstm_finalize_kernel<<<(BATCH * HSZ + 255) / 256, 256, 0, stream>>>(
      out + (size_t)(T_STEPS - 1) * BATCH * HSZ, cst,
      out + (size_t)T_STEPS * BATCH * HSZ,
      out + (size_t)T_STEPS * BATCH * HSZ + (size_t)BATCH * HSZ,
      BATCH * HSZ);
}
